// Decoder_3590592659978
// MI455X (gfx1250) — matrix operand order repaired and checked by reading, not compiled, not run on hardware
//
#include <hip/hip_runtime.h>

typedef __attribute__((ext_vector_type(16))) __bf16 v16bf;
typedef __attribute__((ext_vector_type(8)))  __bf16 v8bf;
typedef __attribute__((ext_vector_type(8)))  float  v8f;

#define V_  16000
#define H_  512
#define T_  128
#define N_  32

// LDS layout for the persistent kernel: 3 matrices x 512 K-rows x (4 gates x 16)
// columns, row stride padded to 72 elements (144B) to spread the 32-lane
// ds_load pattern across all 64 LDS banks.
#define LDS_ROW   72
#define LDS_MAT   (512 * LDS_ROW)
#define LDS_BYTES (3 * LDS_MAT * 2)

__device__ __forceinline__ v8f wmma_bf16(v16bf a, v16bf b, v8f c) {
  // D = A(16x32 bf16) x B(32x16 bf16) + C(16x16 f32)
  return __builtin_amdgcn_wmma_f32_16x16x32_bf16(false, a, false, b, (short)0, c,
                                                 false, false);
}

// A-tile fragment (ISA 16-bit A layout): p -> A[m*K + kt*32 + (lane>>4)*8]
// elements 0..7 = k kb..kb+7 ; elements 8..15 = k kb+16..kb+23
__device__ __forceinline__ v16bf load_a16(const __bf16* p) {
  union { v16bf v; v8bf h[2]; } u;
  u.h[0] = *(const v8bf*)p;
  u.h[1] = *(const v8bf*)(p + 16);
  return u.v;
}

// B fragment from a 16B-aligned (not 32B) location: two 16B halves.
__device__ __forceinline__ v16bf load_b16(const __bf16* p) {
  union { v16bf v; v8bf h[2]; } u;
  u.h[0] = *(const v8bf*)p;
  u.h[1] = *(const v8bf*)(p + 8);
  return u.v;
}

// ---------------------------------------------------------------------------
// Prep kernels
// ---------------------------------------------------------------------------

// Wt[k][n] = (bf16) W[n][k]   (W is rows x cols, Wt is cols x rows, K-major)
__global__ void transpose_to_bf16(const float* __restrict__ W, __bf16* __restrict__ Wt,
                                  int rows, int cols) {
  long long o = (long long)blockIdx.x * blockDim.x + threadIdx.x;
  long long total = (long long)rows * cols;
  if (o >= total) return;
  int k = (int)(o / rows);
  int n = (int)(o % rows);
  Wt[o] = (__bf16)W[(long long)n * cols + k];
}

__global__ void add_bias_vec(const float* __restrict__ a, const float* __restrict__ b,
                             float* __restrict__ o, int n) {
  int i = blockIdx.x * blockDim.x + threadIdx.x;
  if (i < n) o[i] = a[i] + b[i];
}

// Xin[t][n][0:512] = emb[text[n][t]] ; Xin[t][n][512:1024] = values[t][n]
// Also fill Z[t][n][512:1024] = values (Z[...,:512] = h2, written by recurrence)
__global__ void pack_inputs(const float* __restrict__ emb, const int* __restrict__ text,
                            const float* __restrict__ values,
                            __bf16* __restrict__ Xin, __bf16* __restrict__ Z) {
  int tn = blockIdx.x;            // t*N + n
  int t = tn >> 5;
  int n = tn & 31;
  int tok = text[n * T_ + t];
  const float* erow = emb + (long long)tok * H_;
  const float* vrow = values + (long long)tn * H_;
  for (int j = threadIdx.x; j < 1024; j += blockDim.x) {
    float v = (j < 512) ? erow[j] : vrow[j - 512];
    __bf16 bv = (__bf16)v;
    Xin[(long long)tn * 1024 + j] = bv;
    if (j >= 512) Z[(long long)tn * 1024 + j] = bv;
  }
}

// ---------------------------------------------------------------------------
// bf16 WMMA GEMM: one wave per block, 64(M) x 64(N) tile per wave
// (4 M-tiles x 4 N-tiles -> 8 fragment loads : 16 WMMAs per k-step).
// out = A(MxK) * B(KxNn) + bias ;  mode 0: out[m][col] f32 row-major
// mode 1: decoder output: row m = t*32+nb  ->  out[nb][t][col]  (N,T,V)
// ---------------------------------------------------------------------------
__global__ __launch_bounds__(32) void gemm_bf16_64x64(
    const __bf16* __restrict__ A, const __bf16* __restrict__ B,
    const float* __restrict__ bias, float* __restrict__ out,
    int Nn, int K, int mode) {
  const int lane = threadIdx.x;
  const int l15 = lane & 15, lhi = lane >> 4;
  const int baseN = blockIdx.x * 64;
  const int baseM = blockIdx.y * 64;

  v8f acc[4][4];
  const v8f vz = {0.f, 0.f, 0.f, 0.f, 0.f, 0.f, 0.f, 0.f};
#pragma unroll
  for (int i = 0; i < 4; ++i)
#pragma unroll
    for (int j = 0; j < 4; ++j) acc[i][j] = vz;

  const int ksteps = K >> 5;
  for (int kt = 0; kt < ksteps; ++kt) {
    const int kb = (kt << 5) + lhi * 8;
    v16bf a[4];
#pragma unroll
    for (int mt = 0; mt < 4; ++mt)
      a[mt] = load_a16(A + (size_t)(baseM + mt * 16 + l15) * K + kb);
    const int krow = (kt << 5) + lhi * 8;
    const __bf16* brow = B + (size_t)krow * Nn + baseN + l15;
#pragma unroll
    for (int nt = 0; nt < 4; ++nt) {
      v16bf b;
#pragma unroll
      for (int s = 0; s < 16; ++s)
        b[s] = brow[(size_t)((s & 7) + ((s >> 3) << 4)) * Nn + nt * 16];
#pragma unroll
      for (int mt = 0; mt < 4; ++mt)
        acc[mt][nt] = wmma_bf16(a[mt], b, acc[mt][nt]);
    }
  }

#pragma unroll
  for (int mt = 0; mt < 4; ++mt)
#pragma unroll
    for (int nt = 0; nt < 4; ++nt) {
      const int col = baseN + nt * 16 + l15;
      const float bv = bias ? bias[col] : 0.f;
#pragma unroll
      for (int r = 0; r < 8; ++r) {
        const int m = baseM + mt * 16 + lhi * 8 + r;
        const float v = acc[mt][nt][r] + bv;
        if (mode == 0) {
          out[(size_t)m * Nn + col] = v;
        } else {
          const int t = m >> 5, nb = m & 31;
          out[((size_t)nb * T_ + t) * V_ + col] = v;
        }
      }
    }
}

// ---------------------------------------------------------------------------
// Persistent LSTM recurrence. 32 blocks x 1 wave; block hn owns hidden
// columns [hn*16, hn*16+16). Recurrent weight slices cached in LDS once;
// cell state c1/c2 lives in registers for all 128 steps; h1/h2 ping-pong
// through global, synchronized with per-step device-scope atomic barriers.
// ---------------------------------------------------------------------------
__device__ __forceinline__ void grid_barrier(unsigned* cnt) {
  __threadfence();
  if (threadIdx.x == 0)
    __hip_atomic_fetch_add(cnt, 1u, __ATOMIC_ACQ_REL, __HIP_MEMORY_SCOPE_AGENT);
  while (__hip_atomic_load(cnt, __ATOMIC_ACQUIRE, __HIP_MEMORY_SCOPE_AGENT) < (unsigned)N_)
    __builtin_amdgcn_s_sleep(1);
  __threadfence();
}

__global__ __launch_bounds__(32) void lstm_persistent(
    const float* __restrict__ G1,      // (T*N) x 2048 pre-gates, layer 1
    const float* __restrict__ bias2,   // 2048
    const __bf16* __restrict__ Wh1t,   // 512 x 2048 (K-major)
    const __bf16* __restrict__ Wi2t,   // 512 x 2048
    const __bf16* __restrict__ Wh2t,   // 512 x 2048
    __bf16* __restrict__ h1a, __bf16* __restrict__ h1b,   // ping-pong 32x512
    __bf16* __restrict__ h2a, __bf16* __restrict__ h2b,
    __bf16* __restrict__ Z,            // (T*N) x 1024 ; cols 0..511 written here
    unsigned* __restrict__ barriers)   // 2*T_ zeroed counters
{
  extern __shared__ __bf16 lds[];
  const int lane = threadIdx.x;
  const int l15 = lane & 15, lhi = lane >> 4;
  const int hn = blockIdx.x;
  const int hid = hn * 16 + l15;

  // Cache this block's gate-column slices (B-fragment order) in LDS.
  const __bf16* wsrc[3] = {Wh1t, Wi2t, Wh2t};
  for (int mat = 0; mat < 3; ++mat) {
    const __bf16* src = wsrc[mat];
    for (int k = lane; k < 512; k += 32) {
      const __bf16* s = src + (size_t)k * 2048 + hn * 16;
      __bf16* d = lds + mat * LDS_MAT + k * LDS_ROW;
#pragma unroll
      for (int g = 0; g < 4; ++g) {
        union { v16bf v; v8bf h[2]; } u;
        u.v = *(const v16bf*)(s + g * 512);
        *(v8bf*)(d + g * 16) = u.h[0];
        *(v8bf*)(d + g * 16 + 8) = u.h[1];
      }
    }
  }
  __syncthreads();

  // Per-lane cell state (16 of the 32 rows x this lane's hid column).
  float c1r[16], c2r[16];
#pragma unroll
  for (int i = 0; i < 16; ++i) { c1r[i] = 0.f; c2r[i] = 0.f; }

  // Hoist layer-2 bias.
  const float b2i = bias2[hid], b2f = bias2[hid + 512];
  const float b2g = bias2[hid + 1024], b2o = bias2[hid + 1536];

  __bf16* h1ping[2] = {h1a, h1b};
  __bf16* h2ping[2] = {h2a, h2b};
  const v8f vz = {0.f, 0.f, 0.f, 0.f, 0.f, 0.f, 0.f, 0.f};

  for (int t = 0; t < T_; ++t) {
    const __bf16* h1in = h1ping[t & 1];
    __bf16*       h1o  = h1ping[(t + 1) & 1];
    const __bf16* h2in = h2ping[t & 1];
    __bf16*       h2o  = h2ping[(t + 1) & 1];

    // ---------------- layer 1: gates = G1[t] + h1in @ W_hh1^T ----------------
    v8f acc[2][4];
#pragma unroll
    for (int i = 0; i < 2; ++i)
#pragma unroll
      for (int j = 0; j < 4; ++j) acc[i][j] = vz;

    for (int kt = 0; kt < 16; ++kt) {
      const int kb = (kt << 5) + lhi * 8;
      v16bf a0 = load_a16(h1in + (size_t)l15 * 512 + kb);
      v16bf a1 = load_a16(h1in + (size_t)(16 + l15) * 512 + kb);
      const __bf16* brow = lds + 0 * LDS_MAT + ((kt << 5) + lhi * 8) * LDS_ROW + l15;
#pragma unroll
      for (int g = 0; g < 4; ++g) {
        v16bf b;
#pragma unroll
        for (int s = 0; s < 16; ++s)
          b[s] = brow[((s & 7) + ((s >> 3) << 4)) * LDS_ROW + g * 16];
        acc[0][g] = wmma_bf16(a0, b, acc[0][g]);
        acc[1][g] = wmma_bf16(a1, b, acc[1][g]);
      }
    }

#pragma unroll
    for (int mt = 0; mt < 2; ++mt)
#pragma unroll
      for (int r = 0; r < 8; ++r) {
        const int m = mt * 16 + lhi * 8 + r;
        const int idx = mt * 8 + r;
        const float* p = G1 + ((size_t)t * N_ + m) * 2048 + hid;
        const float vi = acc[mt][0][r] + p[0];
        const float vf = acc[mt][1][r] + p[512];
        const float vg = acc[mt][2][r] + p[1024];
        const float vo = acc[mt][3][r] + p[1536];
        const float si = 1.f / (1.f + expf(-vi));
        const float sf = 1.f / (1.f + expf(-vf));
        const float so = 1.f / (1.f + expf(-vo));
        const float cn = sf * c1r[idx] + si * tanhf(vg);
        c1r[idx] = cn;
        h1o[(size_t)m * 512 + hid] = (__bf16)(so * tanhf(cn));
      }

    grid_barrier(barriers + 2 * t);      // h1o visible to all blocks

    // ------- layer 2: gates = bias2 + h1o @ W_ih2^T + h2in @ W_hh2^T -------
#pragma unroll
    for (int i = 0; i < 2; ++i)
#pragma unroll
      for (int j = 0; j < 4; ++j) acc[i][j] = vz;

#pragma unroll
    for (int pass = 0; pass < 2; ++pass) {
      const __bf16* A = pass ? h2in : h1o;
      const int matOff = (pass ? 2 : 1) * LDS_MAT;
      for (int kt = 0; kt < 16; ++kt) {
        const int kb = (kt << 5) + lhi * 8;
        v16bf a0 = load_a16(A + (size_t)l15 * 512 + kb);
        v16bf a1 = load_a16(A + (size_t)(16 + l15) * 512 + kb);
        const __bf16* brow = lds + matOff + ((kt << 5) + lhi * 8) * LDS_ROW + l15;
#pragma unroll
        for (int g = 0; g < 4; ++g) {
          v16bf b;
#pragma unroll
          for (int s = 0; s < 16; ++s)
            b[s] = brow[((s & 7) + ((s >> 3) << 4)) * LDS_ROW + g * 16];
          acc[0][g] = wmma_bf16(a0, b, acc[0][g]);
          acc[1][g] = wmma_bf16(a1, b, acc[1][g]);
        }
      }
    }

#pragma unroll
    for (int mt = 0; mt < 2; ++mt)
#pragma unroll
      for (int r = 0; r < 8; ++r) {
        const int m = mt * 16 + lhi * 8 + r;
        const int idx = mt * 8 + r;
        const float vi = acc[mt][0][r] + b2i;
        const float vf = acc[mt][1][r] + b2f;
        const float vg = acc[mt][2][r] + b2g;
        const float vo = acc[mt][3][r] + b2o;
        const float si = 1.f / (1.f + expf(-vi));
        const float sf = 1.f / (1.f + expf(-vf));
        const float so = 1.f / (1.f + expf(-vo));
        const float cn = sf * c2r[idx] + si * tanhf(vg);
        c2r[idx] = cn;
        const __bf16 hv = (__bf16)(so * tanhf(cn));
        h2o[(size_t)m * 512 + hid] = hv;
        Z[((size_t)t * N_ + m) * 1024 + hid] = hv;
      }

    grid_barrier(barriers + 2 * t + 1);  // h2o / Z visible before next step
  }
}

// ---------------------------------------------------------------------------
extern "C" void kernel_launch(void* const* d_in, const int* in_sizes, int n_in,
                              void* d_out, int out_size, void* d_ws, size_t ws_size,
                              hipStream_t stream) {
  (void)in_sizes; (void)n_in; (void)out_size; (void)ws_size;
  const float* values = (const float*)d_in[1];
  const int*   text   = (const int*)d_in[3];
  const float* emb    = (const float*)d_in[4];
  const float* W_ih1  = (const float*)d_in[5];
  const float* W_hh1  = (const float*)d_in[6];
  const float* b_ih1  = (const float*)d_in[7];
  const float* b_hh1  = (const float*)d_in[8];
  const float* W_ih2  = (const float*)d_in[9];
  const float* W_hh2  = (const float*)d_in[10];
  const float* b_ih2  = (const float*)d_in[11];
  const float* b_hh2  = (const float*)d_in[12];
  const float* W_out  = (const float*)d_in[13];
  const float* b_out  = (const float*)d_in[14];
  float* out = (float*)d_out;

  char* ws = (char*)d_ws;
  size_t off = 0;
  auto take = [&](size_t bytes) { char* p = ws + off; off += (bytes + 255) & ~(size_t)255; return p; };

  __bf16*   W1t   = (__bf16*)take((size_t)1024 * 2048 * 2);   // (K=1024) x (4H=2048)
  __bf16*   Wh1t  = (__bf16*)take((size_t)512  * 2048 * 2);
  __bf16*   Wi2t  = (__bf16*)take((size_t)512  * 2048 * 2);
  __bf16*   Wh2t  = (__bf16*)take((size_t)512  * 2048 * 2);
  __bf16*   Wot   = (__bf16*)take((size_t)1024 * V_   * 2);   // (1024) x (16000)
  float*    bias1 = (float*)take((size_t)2048 * 4);
  float*    bias2 = (float*)take((size_t)2048 * 4);
  __bf16*   Xin   = (__bf16*)take((size_t)T_ * N_ * 1024 * 2);
  float*    G1    = (float*)take((size_t)T_ * N_ * 2048 * 4);
  __bf16*   Z     = (__bf16*)take((size_t)T_ * N_ * 1024 * 2);
  __bf16*   h1a   = (__bf16*)take((size_t)N_ * 512 * 2);
  __bf16*   h1b   = (__bf16*)take((size_t)N_ * 512 * 2);
  __bf16*   h2a   = (__bf16*)take((size_t)N_ * 512 * 2);
  __bf16*   h2b   = (__bf16*)take((size_t)N_ * 512 * 2);
  unsigned* bars  = (unsigned*)take((size_t)2 * T_ * 4);

  // zero initial state + barrier counters (every call; harness does not re-poison)
  hipMemsetAsync(h1a, 0, (size_t)N_ * 512 * 2, stream);
  hipMemsetAsync(h2a, 0, (size_t)N_ * 512 * 2, stream);
  hipMemsetAsync(bars, 0, (size_t)2 * T_ * 4, stream);

  // prep: biases, K-major bf16 weights, packed inputs
  add_bias_vec<<<8, 256, 0, stream>>>(b_ih1, b_hh1, bias1, 2048);
  add_bias_vec<<<8, 256, 0, stream>>>(b_ih2, b_hh2, bias2, 2048);
  transpose_to_bf16<<<(2048 * 1024) / 256, 256, 0, stream>>>(W_ih1, W1t, 2048, 1024);
  transpose_to_bf16<<<(2048 * 512) / 256, 256, 0, stream>>>(W_hh1, Wh1t, 2048, 512);
  transpose_to_bf16<<<(2048 * 512) / 256, 256, 0, stream>>>(W_ih2, Wi2t, 2048, 512);
  transpose_to_bf16<<<(2048 * 512) / 256, 256, 0, stream>>>(W_hh2, Wh2t, 2048, 512);
  transpose_to_bf16<<<(V_ * 1024) / 256, 256, 0, stream>>>(W_out, Wot, V_, 1024);
  pack_inputs<<<T_ * N_, 256, 0, stream>>>(emb, text, values, Xin, Z);

  // precompute input-part pre-gates for LSTM1:  G1 = Xin @ W_ih1^T + bias1
  gemm_bf16_64x64<<<dim3(2048 / 64, (T_ * N_) / 64), 32, 0, stream>>>(
      Xin, W1t, bias1, G1, 2048, 1024, 0);

  // whole sequential recurrence in ONE persistent kernel
  lstm_persistent<<<32, 32, LDS_BYTES, stream>>>(
      G1, bias2, Wh1t, Wi2t, Wh2t, h1a, h1b, h2a, h2b, Z, bars);

  // output projection:  out[n][t][:] = [h2 | v] @ W_out^T + b_out
  gemm_bf16_64x64<<<dim3(V_ / 64, (T_ * N_) / 64), 32, 0, stream>>>(
      Z, Wot, b_out, out, V_, 1024, 1);
}
